// Att_Baseline_44178033607249
// MI455X (gfx1250) — matrix operand order repaired and checked by reading, not compiled, not run on hardware
//
#include <hip/hip_runtime.h>

#define DEV __device__ __forceinline__
#define USE_ASYNC_A_STAGE 1

typedef __attribute__((ext_vector_type(16))) __bf16         bf16x16;
typedef __attribute__((ext_vector_type(8)))  float          f32x8;
typedef __attribute__((ext_vector_type(8)))  unsigned short us8;
typedef __attribute__((ext_vector_type(16))) unsigned short us16;

union AFrag { us8  h[2]; bf16x16 v; };
union BFrag { us16 u;    bf16x16 v; };

DEV unsigned short f2bf(float f) {
    unsigned int u = __float_as_uint(f);
    return (unsigned short)((u + 0x7FFFu + ((u >> 16) & 1u)) >> 16);
}
DEV float sigm(float x) { return 1.0f / (1.0f + __expf(-x)); }

// A fragment: 16x32 bf16 (MxK). lane<16: m=lane, K 0..15;
// lane>=16: m=lane-16, K 16..31.
DEV bf16x16 load_a(const unsigned short* A, int lda, int m0, int k0, int lane) {
    int half = lane >> 4, mr = lane & 15;
    const unsigned short* p = A + (size_t)(m0 + mr) * lda + (k0 + half * 16);
    AFrag f;
    f.h[0] = *(const us8*)(p);
    f.h[1] = *(const us8*)(p + 8);
    return f.v;
}
// Same A fragment sourced from an LDS-staged 16x512 tile.
DEV bf16x16 load_a_sh(const unsigned short* As, int k0, int lane) {
    int half = lane >> 4, mr = lane & 15;
    const unsigned short* p = As + mr * 512 + (k0 + half * 16);
    AFrag f;
    f.h[0] = *(const us8*)(p);
    f.h[1] = *(const us8*)(p + 8);
    return f.v;
}
// B fragment: 32x16 bf16 (KxN), B stored [N,K] row-major. lane holds n=lane&15,
// 16 contiguous K at k0 + (lane>>4)*16.
DEV bf16x16 load_b(const unsigned short* Bm, int ldb, int n0, int k0, int lane) {
    const unsigned short* p = Bm + (size_t)(n0 + (lane & 15)) * ldb + (k0 + (lane >> 4) * 16);
    BFrag f;
    f.u = *(const us16*)p;
    return f.v;
}
DEV f32x8 wmma_bf16(bf16x16 a, bf16x16 b, f32x8 c) {
    return __builtin_amdgcn_wmma_f32_16x16x32_bf16(false, a, false, b, (short)0, c, false, false);
}

// ---------------------------------------------------------------------------
// Generic NT GEMM: C[M,N] = A[M,K] * B[N,K]^T (+bias[n]). One wave computes a
// 16 x (16*NT) tile: NT independent accumulators (back-to-back WMMA issue) and
// software-pipelined fragment loads (k+32 loaded while k is computed).
// outmode==1 remaps row r=t*64+b -> out[(b*39+t)*ldc + col].
// ---------------------------------------------------------------------------
template <int NT>
__global__ __launch_bounds__(256) void gemm_bf16_nt(
    const unsigned short* __restrict__ A, const unsigned short* __restrict__ Bm,
    const float* __restrict__ bias, float* __restrict__ C,
    unsigned short* __restrict__ Cbf,
    int M, int N, int K, int lda, int ldb, int ldc, int outmode)
{
    int gtid = blockIdx.x * 256 + threadIdx.x;
    int wave = gtid >> 5;
    int lane = gtid & 31;
    int tilesN = N / (16 * NT);
    int tilesM = M >> 4;
    if (wave >= tilesM * tilesN) return;   // wave-uniform: EXEC all-ones for WMMA
    int mt = wave / tilesN;
    int nt = wave - mt * tilesN;
    int m0 = mt << 4, n0 = nt * 16 * NT;

    f32x8 acc[NT];
#pragma unroll
    for (int j = 0; j < NT; ++j) acc[j] = (f32x8){};

    bf16x16 a = load_a(A, lda, m0, 0, lane);
    bf16x16 bfr[NT];
#pragma unroll
    for (int j = 0; j < NT; ++j) bfr[j] = load_b(Bm, ldb, n0 + j * 16, 0, lane);

    for (int k0 = 0; k0 < K; k0 += 32) {
        bf16x16 an = a;
        bf16x16 bn[NT];
#pragma unroll
        for (int j = 0; j < NT; ++j) bn[j] = bfr[j];
        if (k0 + 32 < K) {
            an = load_a(A, lda, m0, k0 + 32, lane);
#pragma unroll
            for (int j = 0; j < NT; ++j) {
                bn[j] = load_b(Bm, ldb, n0 + j * 16, k0 + 32, lane);
                if (k0 + 64 < K)
                    __builtin_prefetch(Bm + (size_t)(n0 + j * 16 + (lane & 15)) * ldb + k0 + 64, 0, 1);
            }
        }
#pragma unroll
        for (int j = 0; j < NT; ++j) acc[j] = wmma_bf16(a, bfr[j], acc[j]);
        a = an;
#pragma unroll
        for (int j = 0; j < NT; ++j) bfr[j] = bn[j];
    }

    int half = lane >> 4;
#pragma unroll
    for (int j = 0; j < NT; ++j) {
        int col = n0 + j * 16 + (lane & 15);
        float bv = bias ? bias[col] : 0.0f;
#pragma unroll
        for (int v = 0; v < 8; ++v) {
            int row = m0 + v + 8 * half;
            float val = acc[j][v] + bv;
            if (outmode == 1) {
                int b_ = row & 63, t_ = row >> 6;
                C[(size_t)(b_ * 39 + t_) * ldc + col] = val;
            } else {
                if (C)   C[(size_t)row * ldc + col]   = val;
                if (Cbf) Cbf[(size_t)row * ldc + col] = f2bf(val);
            }
        }
    }
}

// ---------------------------------------------------------------------------
// Fused LSTM step: stage h (16x512 bf16 A-tile) into LDS via async DMA
// (ASYNCcnt path), gates = h @ Whh^T with WMMA (K=512) staged in LDS, then
// gates += gx[t] (+extra), pointwise cell update, write bf16 h for next step.
// Grid = ndir*4 blocks of 256 threads; block = (dir, m-tile of 16 rows).
// ---------------------------------------------------------------------------
__global__ __launch_bounds__(256) void lstm_step_kernel(
    const unsigned short* __restrict__ Whh_f, const unsigned short* __restrict__ Whh_b,
    const float* __restrict__ gxt_f, const float* __restrict__ gxt_b, long gx_bstride,
    const float* __restrict__ extra,
    unsigned short* __restrict__ hbf_f, unsigned short* __restrict__ hbf_b,
    float* __restrict__ c_f, float* __restrict__ c_b,
    float* __restrict__ hout_f, float* __restrict__ hout_b,
    unsigned short* __restrict__ hbfout)
{
    extern __shared__ float g_lds[];            // 16 x 2048 f32 gate strip (128 KB)
    __shared__ unsigned short a_stage[16 * 512]; // 16 KB staged A tile
    int dir = blockIdx.x >> 2;
    int mt  = blockIdx.x & 3;
    int m0  = mt << 4;

    const unsigned short* Whh = dir ? Whh_b : Whh_f;
    const float* gxt          = dir ? gxt_b : gxt_f;
    unsigned short* hbf       = dir ? hbf_b : hbf_f;
    float* cbuf               = dir ? c_b   : c_f;
    float* hout               = dir ? hout_b : hout_f;

    int wave  = threadIdx.x >> 5;
    int lane  = threadIdx.x & 31;
    int wbase = wave << 8;                      // 256 gate columns per wave

    // ---- stage A tile (rows m0..m0+15 of h) into LDS ----
#if USE_ASYNC_A_STAGE
    {
        // Generic LDS pointer low 32 bits == wave-relative LDS byte address
        // (ISA 10.2 aperture rule).
        unsigned lbase = (unsigned)(size_t)(const void*)a_stage;
#pragma unroll
        for (int it = 0; it < 4; ++it) {
            int c   = it * 256 + threadIdx.x;   // 16B chunk id (1024 total = 16 KB)
            int row = c >> 6;                   // 0..15
            int cof = c & 63;                   // 16B chunk within 1 KB row
            const unsigned short* gp = hbf + (size_t)(m0 + row) * 512 + cof * 8;
            unsigned la = lbase + (unsigned)(row * 1024 + cof * 16);
            unsigned long long ga = (unsigned long long)(size_t)gp;
            asm volatile("global_load_async_to_lds_b128 %0, %1, off"
                         :: "v"(la), "v"(ga) : "memory");
        }
        asm volatile("s_wait_asynccnt 0x0" ::: "memory");
    }
#else
#pragma unroll
    for (int it = 0; it < 4; ++it) {
        int c   = it * 256 + threadIdx.x;
        int row = c >> 6;
        int cof = c & 63;
        *(us8*)(a_stage + row * 512 + cof * 8) =
            *(const us8*)(hbf + (size_t)(m0 + row) * 512 + cof * 8);
    }
#endif
    __syncthreads();

    f32x8 z = {};
#pragma unroll
    for (int hn = 0; hn < 2; ++hn) {
        f32x8 acc[8];
#pragma unroll
        for (int j = 0; j < 8; ++j) acc[j] = z;
        for (int k0 = 0; k0 < 512; k0 += 32) {
            bf16x16 a = load_a_sh(a_stage, k0, lane);
#pragma unroll
            for (int j = 0; j < 8; ++j) {
                bf16x16 b = load_b(Whh, 512, wbase + hn * 128 + j * 16, k0, lane);
                acc[j] = wmma_bf16(a, b, acc[j]);
            }
        }
        int half = lane >> 4;
#pragma unroll
        for (int j = 0; j < 8; ++j) {
            int col = wbase + hn * 128 + j * 16 + (lane & 15);
#pragma unroll
            for (int v = 0; v < 8; ++v) {
                int r = v + 8 * half;
                g_lds[r * 2048 + col] = acc[j][v];
            }
        }
    }
    __syncthreads();

    // Pointwise: 16 rows x 512 cols = 8192 elements, 32 per thread.
    for (int e = 0; e < 32; ++e) {
        int lin = e * 256 + threadIdx.x;
        int r = lin >> 9;                       // local row 0..15
        int n = lin & 511;                      // hidden index
        int brow = m0 + r;                      // batch index
        size_t gb = (size_t)brow * gx_bstride;
        float xi = g_lds[r * 2048 + n        ] + gxt[gb + n        ];
        float xf = g_lds[r * 2048 + 512  + n ] + gxt[gb + 512  + n ];
        float xg = g_lds[r * 2048 + 1024 + n ] + gxt[gb + 1024 + n ];
        float xo = g_lds[r * 2048 + 1536 + n ] + gxt[gb + 1536 + n ];
        if (extra) {
            size_t eb = (size_t)brow * 2048;
            xi += extra[eb + n];         xf += extra[eb + 512 + n];
            xg += extra[eb + 1024 + n];  xo += extra[eb + 1536 + n];
        }
        size_t idx = (size_t)brow * 512 + n;
        float cc = sigm(xf) * cbuf[idx] + sigm(xi) * tanhf(xg);
        float hh = sigm(xo) * tanhf(cc);
        cbuf[idx] = cc;
        hbf[idx]  = f2bf(hh);
        if (hout)   hout[idx]   = hh;
        if (hbfout) hbfout[idx] = f2bf(hh);
    }
}

// ------------------------------ small helpers ------------------------------
__global__ void cvt_bf16_kernel(const float* __restrict__ in,
                                unsigned short* __restrict__ out, long n) {
    long i = (long)blockIdx.x * 256 + threadIdx.x;
    if (i < n) out[i] = f2bf(in[i]);
}
__global__ void addvec_kernel(const float* __restrict__ a, const float* __restrict__ b,
                              float* __restrict__ o, int n) {
    int i = blockIdx.x * 256 + threadIdx.x;
    if (i < n) o[i] = a[i] + b[i];
}
__global__ void zero_f32_kernel(float* __restrict__ p, long n) {
    long i = (long)blockIdx.x * 256 + threadIdx.x;
    if (i < n) p[i] = 0.0f;
}
__global__ void gather_emb_kernel(const float* __restrict__ emb,
                                  const int* __restrict__ targets,
                                  unsigned short* __restrict__ embt) {
    long i = (long)blockIdx.x * 256 + threadIdx.x;
    if (i >= (long)39 * 64 * 512) return;
    int j = (int)(i & 511);
    int r = (int)(i >> 9);                      // r = t*64 + b
    int b = r & 63, t = r >> 6;
    int tok = targets[b * 40 + t];
    embt[i] = f2bf(emb[(size_t)tok * 512 + j]);
}
// ctx[b, :] = sum_t enc[b,t,:] (softmax over singleton dim == ones, exactly)
__global__ void ctx_kernel(const float* __restrict__ hf, const float* __restrict__ hb,
                           unsigned short* __restrict__ ctx) {
    int i = blockIdx.x * 256 + threadIdx.x;
    if (i >= 64 * 1024) return;
    int j2 = i & 1023, b = i >> 10;
    const float* src = (j2 < 512) ? hf : hb;
    int j = j2 & 511;
    float s = 0.0f;
    for (int t = 0; t < 40; ++t) s += src[(size_t)t * 64 * 512 + (size_t)b * 512 + j];
    ctx[i] = f2bf(s);
}

static inline void gemm_launch(hipStream_t s, const unsigned short* A, const unsigned short* Bm,
                               const float* bias, float* C, unsigned short* Cbf,
                               int M, int N, int K, int lda, int ldb, int ldc, int outmode) {
    int NT = (N % 64 == 0) ? 4 : ((N % 32 == 0) ? 2 : 1);
    long waves  = (long)(M >> 4) * (N / (16 * NT));
    int  blocks = (int)((waves * 32 + 255) / 256);
    switch (NT) {
    case 4:  gemm_bf16_nt<4><<<blocks, 256, 0, s>>>(A, Bm, bias, C, Cbf, M, N, K, lda, ldb, ldc, outmode); break;
    case 2:  gemm_bf16_nt<2><<<blocks, 256, 0, s>>>(A, Bm, bias, C, Cbf, M, N, K, lda, ldb, ldc, outmode); break;
    default: gemm_bf16_nt<1><<<blocks, 256, 0, s>>>(A, Bm, bias, C, Cbf, M, N, K, lda, ldb, ldc, outmode); break;
    }
}

extern "C" void kernel_launch(void* const* d_in, const int* in_sizes, int n_in,
                              void* d_out, int out_size, void* d_ws, size_t ws_size,
                              hipStream_t stream)
{
    (void)in_sizes; (void)n_in; (void)out_size; (void)ws_size;

    const float* feats   = (const float*)d_in[0];
    const float* feat_W  = (const float*)d_in[1];
    const float* feat_b  = (const float*)d_in[2];
    const float* Wih_f   = (const float*)d_in[3];
    const float* Whh_f   = (const float*)d_in[4];
    const float* bih_f   = (const float*)d_in[5];
    const float* bhh_f   = (const float*)d_in[6];
    const float* Wih_b   = (const float*)d_in[7];
    const float* Whh_b   = (const float*)d_in[8];
    const float* bih_b   = (const float*)d_in[9];
    const float* bhh_b   = (const float*)d_in[10];
    const float* emb     = (const float*)d_in[11];
    const float* Wih_d   = (const float*)d_in[12];
    const float* Whh_d   = (const float*)d_in[13];
    const float* bih_d   = (const float*)d_in[14];
    const float* bhh_d   = (const float*)d_in[15];
    // d_in[16..20]: attE_W/attE_b/attP_W/attP_b/attA_w — mathematically dead
    const float* out_W   = (const float*)d_in[21];
    const float* out_b   = (const float*)d_in[22];
    const int*   targets = (const int*)d_in[23];
    float* out = (float*)d_out;

    // ---------------- workspace carve-up (256B aligned) ----------------
    char* ws = (char*)d_ws;
    size_t off = 0;
    auto alloc = [&](size_t bytes) -> char* {
        char* p = ws + off;
        off = (off + bytes + 255) & ~(size_t)255;
        return p;
    };
    unsigned short* feats_bf = (unsigned short*)alloc(5242880L * 2);  // [2560,2048]
    unsigned short* featW_bf = (unsigned short*)alloc(1048576L * 2);  // [512,2048]
    unsigned short* Wihf_bf  = (unsigned short*)alloc(1048576L * 2);  // [2048,512]
    unsigned short* Whhf_bf  = (unsigned short*)alloc(1048576L * 2);
    unsigned short* Wihb_bf  = (unsigned short*)alloc(1048576L * 2);
    unsigned short* Whhb_bf  = (unsigned short*)alloc(1048576L * 2);
    unsigned short* Wd_bf    = (unsigned short*)alloc(3145728L * 2);  // [2048,1536]
    unsigned short* Whhd_bf  = (unsigned short*)alloc(1048576L * 2);
    unsigned short* outW_bf  = (unsigned short*)alloc(10240000L * 2); // [20000,512]
    unsigned short* x_bf     = (unsigned short*)alloc(1310720L * 2);  // [2560,512]
    float*          gx_f     = (float*)alloc(5242880L * 4);           // [2560,2048]
    float*          gx_b     = (float*)alloc(5242880L * 4);
    float*          gxd      = (float*)alloc(5111808L * 4);           // [2496,2048]
    float*          hf_buf   = (float*)alloc(1310720L * 4);           // [40,64,512]
    float*          hb_buf   = (float*)alloc(1310720L * 4);
    unsigned short* embt_bf  = (unsigned short*)alloc(1277952L * 2);  // [2496,512]
    unsigned short* Hs_bf    = (unsigned short*)alloc(1277952L * 2);  // [2496,512]
    unsigned short* ctx_bf   = (unsigned short*)alloc(65536L * 2);    // [64,1024]
    float*          dctxb    = (float*)alloc(131072L * 4);            // [64,2048]
    float*          bsum_f   = (float*)alloc(2048L * 4);
    float*          bsum_b   = (float*)alloc(2048L * 4);
    float*          bsum_d   = (float*)alloc(2048L * 4);
    // contiguous zero-initialized state: c (3x f32) then h_bf16 (3x)
    float*          c_f      = (float*)alloc(32768L * 4);
    float*          c_b      = (float*)alloc(32768L * 4);
    float*          c_d      = (float*)alloc(32768L * 4);
    unsigned short* hbf_f    = (unsigned short*)alloc(32768L * 2);
    unsigned short* hbf_b    = (unsigned short*)alloc(32768L * 2);
    unsigned short* hbf_d    = (unsigned short*)alloc(32768L * 2);

    auto cvt = [&](const float* src, unsigned short* dst, long n) {
        cvt_bf16_kernel<<<(int)((n + 255) / 256), 256, 0, stream>>>(src, dst, n);
    };

    // ------------- 1. precision conversion / gathers / init -------------
    cvt(feats,  feats_bf, 5242880L);
    cvt(feat_W, featW_bf, 1048576L);
    cvt(Wih_f,  Wihf_bf,  1048576L);
    cvt(Whh_f,  Whhf_bf,  1048576L);
    cvt(Wih_b,  Wihb_bf,  1048576L);
    cvt(Whh_b,  Whhb_bf,  1048576L);
    cvt(Wih_d,  Wd_bf,    3145728L);
    cvt(Whh_d,  Whhd_bf,  1048576L);
    cvt(out_W,  outW_bf,  10240000L);
    addvec_kernel<<<8, 256, 0, stream>>>(bih_f, bhh_f, bsum_f, 2048);
    addvec_kernel<<<8, 256, 0, stream>>>(bih_b, bhh_b, bsum_b, 2048);
    addvec_kernel<<<8, 256, 0, stream>>>(bih_d, bhh_d, bsum_d, 2048);
    gather_emb_kernel<<<(39 * 64 * 512 + 255) / 256, 256, 0, stream>>>(emb, targets, embt_bf);
    // zero c_f..c_d (3*32768 f32) + hbf_f..hbf_d (3*32768 bf16 == 49152 f32)
    zero_f32_kernel<<<((3 * 32768 + 3 * 16384) + 255) / 256, 256, 0, stream>>>(
        c_f, 3L * 32768 + 3L * 16384);

    // ------------- 2. batched input-side GEMMs -------------
    // x = feats @ feat_W^T + feat_b  (bf16 result only)
    gemm_launch(stream, feats_bf, featW_bf, feat_b, nullptr, x_bf,
                2560, 512, 2048, 2048, 2048, 512, 0);
    // gx = x @ Wih^T + (bih+bhh), both directions, all timesteps
    gemm_launch(stream, x_bf, Wihf_bf, bsum_f, gx_f, nullptr,
                2560, 2048, 512, 512, 512, 2048, 0);
    gemm_launch(stream, x_bf, Wihb_bf, bsum_b, gx_b, nullptr,
                2560, 2048, 512, 512, 512, 2048, 0);
    // gxd = emb_t @ Wd_e^T (cols 0..511 of Wih_d); biases folded into dctxb
    gemm_launch(stream, embt_bf, Wd_bf, nullptr, gxd, nullptr,
                2496, 2048, 512, 512, 1536, 2048, 0);

    // ------------- 3. encoder bi-LSTM: fwd+bwd fused per step -------------
    for (int t = 0; t < 40; ++t) {
        int tb = 39 - t;
        lstm_step_kernel<<<8, 256, 16 * 2048 * 4, stream>>>(
            Whhf_bf, Whhb_bf,
            gx_f + (size_t)t * 2048, gx_b + (size_t)tb * 2048, 40L * 2048,
            nullptr,
            hbf_f, hbf_b, c_f, c_b,
            hf_buf + (size_t)t * 64 * 512, hb_buf + (size_t)tb * 64 * 512,
            nullptr);
    }

    // ------------- 4. constant context + decoder input bias -------------
    ctx_kernel<<<(64 * 1024 + 255) / 256, 256, 0, stream>>>(hf_buf, hb_buf, ctx_bf);
    // dctxb = ctx @ Wd_c^T + (bih_d+bhh_d); Wd_c = cols 512..1535 of Wih_d
    gemm_launch(stream, ctx_bf, Wd_bf + 512, bsum_d, dctxb, nullptr,
                64, 2048, 1024, 1024, 1536, 2048, 0);

    // ------------- 5. decoder LSTM steps -------------
    for (int t = 0; t < 39; ++t) {
        lstm_step_kernel<<<4, 256, 16 * 2048 * 4, stream>>>(
            Whhd_bf, nullptr,
            gxd + (size_t)t * 64 * 2048, nullptr, 2048L,
            dctxb,
            hbf_d, nullptr, c_d, nullptr,
            nullptr, nullptr,
            Hs_bf + (size_t)t * 64 * 512);
    }

    // ------------- 6. batched output projection (dominant 51 GFLOP) -------------
    // probs[b,t,:] = Hs[t*64+b,:] @ out_W^T + out_b, permuted epilogue
    gemm_launch(stream, Hs_bf, outW_bf, out_b, out, nullptr,
                2496, 20000, 512, 512, 512, 20000, 1);
}
